// LocalAttention_62354335203604
// MI455X (gfx1250) — matrix operand order repaired and checked by reading, not compiled, not run on hardware
//
#include <hip/hip_runtime.h>
#include <hip/hip_bf16.h>

// ---------------------------------------------------------------------------
// Types for CDNA5 WMMA (wave32): v_wmma_f32_16x16x32_bf16
// ---------------------------------------------------------------------------
typedef __attribute__((ext_vector_type(16))) __bf16 v16bf;
typedef __attribute__((ext_vector_type(8)))  float  v8f;
typedef unsigned short u16t;
typedef __attribute__((ext_vector_type(8)))  u16t v8u;
typedef __attribute__((ext_vector_type(16))) u16t v16u;

// Exact pointer types for __builtin_amdgcn_global_load_async_to_lds_b128:
// (v4i addrspace(1)* src, v4i addrspace(3)* dst, imm offset, imm cpol)
typedef int v4i __attribute__((vector_size(16)));
typedef __attribute__((address_space(1))) v4i* as1_v4i_p;
typedef __attribute__((address_space(3))) v4i* as3_v4i_p;

#if defined(__has_builtin)
#if __has_builtin(__builtin_amdgcn_global_load_async_to_lds_b128)
#define HAVE_ASYNC_LDS 1
#endif
#endif
#ifndef HAVE_ASYNC_LDS
#define HAVE_ASYNC_LDS 0
#endif

#if defined(__has_builtin)
#if __has_builtin(__builtin_amdgcn_s_wait_asynccnt)
#define WAIT_ASYNC() __builtin_amdgcn_s_wait_asynccnt(0)
#else
#define WAIT_ASYNC() asm volatile("s_wait_asynccnt 0x0" ::: "memory")
#endif
#else
#define WAIT_ASYNC() asm volatile("s_wait_asynccnt 0x0" ::: "memory")
#endif

__device__ __forceinline__ u16t f32_to_bf16(float f) {
    // round-to-nearest-even fp32 -> bf16
    unsigned u = __builtin_bit_cast(unsigned, f);
    u += 0x7FFFu + ((u >> 16) & 1u);
    return (u16t)(u >> 16);
}

__device__ __forceinline__ v8f wmma_bf16(v16bf a, v16bf b, v8f c) {
    return __builtin_amdgcn_wmma_f32_16x16x32_bf16(
        false, a, false, b, (short)0, c, false, false);
}

__device__ __forceinline__ v16bf cat16(v8u lo, v8u hi) {
    return __builtin_bit_cast(v16bf,
        __builtin_shufflevector(lo, hi,
            0, 1, 2, 3, 4, 5, 6, 7, 8, 9, 10, 11, 12, 13, 14, 15));
}

// A fragment (16-bit A 16x32, ISA 7.12.2): lane-half h takes K [8h,8h+8) U [16+8h,24+8h)
__device__ __forceinline__ v16bf frag_a(const u16t* row, int h) {
    return cat16(*(const v8u*)(row + 8 * h), *(const v8u*)(row + 16 + 8 * h));
}
// B fragment (16-bit B 32x16): lane-half h takes K [8h,8h+8) U [16+8h,24+8h)
__device__ __forceinline__ v16bf frag_b(const u16t* row, int h) {
    return cat16(*(const v8u*)(row + 8 * h), *(const v8u*)(row + 16 + 8 * h));
}

// ---------------------------------------------------------------------------
// Prep: elementwise fp32 -> bf16 (vectorized x4)
// ---------------------------------------------------------------------------
__launch_bounds__(256)
__global__ void cvt_bf16(const float* __restrict__ in, u16t* __restrict__ out) {
    const long long i = 4LL * ((long long)blockIdx.x * blockDim.x + threadIdx.x);
    float4 v = *(const float4*)(in + i);
    uint2 p;
    p.x = (unsigned)f32_to_bf16(v.x) | ((unsigned)f32_to_bf16(v.y) << 16);
    p.y = (unsigned)f32_to_bf16(v.z) | ((unsigned)f32_to_bf16(v.w) << 16);
    *(uint2*)(out + i) = p;
}

// Prep: transpose-convert weights: in [K][N] fp32 -> outT [N][K] bf16
__launch_bounds__(256)
__global__ void cvtT_bf16(const float* __restrict__ in, u16t* __restrict__ outT,
                          int K, int N) {
    const long long idx = (long long)blockIdx.x * blockDim.x + threadIdx.x;
    const int k = (int)(idx & (K - 1));     // K is a power of two (512)
    const int n = (int)(idx >> 9);          // idx / 512
    outT[(long long)n * K + k] = f32_to_bf16(in[(long long)k * N + n]);
}

// ---------------------------------------------------------------------------
// WMMA GEMM: C[M x N] fp32 = A[M x K] bf16 (row-major) * Bt[N x K] bf16
// (N-major, i.e. pre-transposed B). Block tile 64x64, TK=64, 8 waves.
// Wave computes 16x32 as two accumulators; 4 WMMAs per k-step.
// Double-buffered LDS with ONE barrier per k-step.
//   - async path: GLOBAL_LOAD_ASYNC_TO_LDS_B128 DMA staging (ASYNCcnt),
//     next tile streamed into the other buffer while WMMAs run.
//   - fallback:   global->regs->LDS double buffer (proven path).
// Plus global_prefetch_b8 L2 prefetch of the tile after next.
// Requires M%64==0, N%64==0, K%128==0.
// ---------------------------------------------------------------------------
#define TM 64
#define TN 64
#define TK 64
#define LDSTR (TK + 8)   // 72 elems = 144B rows: 16B-aligned, staggers banks

__launch_bounds__(256, 2)
__global__ void wmma_gemm_bf16t(const u16t* __restrict__ A,
                                const u16t* __restrict__ Bt,
                                float* __restrict__ C,
                                int N, int Kdim) {
    __shared__ __attribute__((aligned(16))) u16t As[2][TM][LDSTR];
    __shared__ __attribute__((aligned(16))) u16t Bs[2][TN][LDSTR];

    const int tid  = threadIdx.x;
    const int lane = tid & 31;
    const int wv   = tid >> 5;
    const long long bm = (long long)blockIdx.x * TM;
    const long long bn = (long long)blockIdx.y * TN;
    const int wm = (wv & 3) * 16;   // 4 wave-rows x 16
    const int wn = (wv >> 2) * 32;  // 2 wave-cols x 32
    const int h  = lane >> 4;
    const int r  = lane & 15;

    // staging map: 4 threads per 64-elem row, 16 bf16 (32B) per thread
    const int srow = tid >> 2;
    const int soff = (tid & 3) * 16;
    const u16t* agp = A  + (bm + srow) * (long long)Kdim + soff;
    const u16t* bgp = Bt + (bn + srow) * (long long)Kdim + soff;

    v8f acc0 = {0.f, 0.f, 0.f, 0.f, 0.f, 0.f, 0.f, 0.f};
    v8f acc1 = acc0;

    const int S = Kdim / TK;   // 8 for K=512

#define COMPUTE(CUR)                                                     \
    do {                                                                 \
        _Pragma("unroll")                                                \
        for (int s = 0; s < 2; ++s) {                                    \
            v16bf af  = frag_a(&As[CUR][wm + r][32 * s], h);             \
            v16bf bf0 = frag_b(&Bs[CUR][wn + r][32 * s], h);             \
            v16bf bf1 = frag_b(&Bs[CUR][wn + 16 + r][32 * s], h);        \
            acc0 = wmma_bf16(af, bf0, acc0);                             \
            acc1 = wmma_bf16(af, bf1, acc1);                             \
        }                                                                \
    } while (0)

#if HAVE_ASYNC_LDS
    // ---------------- async DMA staging path ----------------
#define AFETCH(K0, BUF)                                                  \
    do {                                                                 \
        __builtin_amdgcn_global_load_async_to_lds_b128(                  \
            (as1_v4i_p)(agp + (K0)),                                     \
            (as3_v4i_p)&As[BUF][srow][soff], 0, 0);                      \
        __builtin_amdgcn_global_load_async_to_lds_b128(                  \
            (as1_v4i_p)(agp + (K0) + 8),                                 \
            (as3_v4i_p)&As[BUF][srow][soff + 8], 0, 0);                  \
        __builtin_amdgcn_global_load_async_to_lds_b128(                  \
            (as1_v4i_p)(bgp + (K0)),                                     \
            (as3_v4i_p)&Bs[BUF][srow][soff], 0, 0);                      \
        __builtin_amdgcn_global_load_async_to_lds_b128(                  \
            (as1_v4i_p)(bgp + (K0) + 8),                                 \
            (as3_v4i_p)&Bs[BUF][srow][soff + 8], 0, 0);                  \
    } while (0)

    AFETCH(0, 0);
    WAIT_ASYNC();
    __syncthreads();

    for (int step = 0; step < S; ++step) {
        const int cur = step & 1;
        if (step + 1 < S) AFETCH((step + 1) * TK, cur ^ 1);  // DMA next tile
        if (step + 2 < S) {                                  // L2 prefetch
            __builtin_prefetch(agp + (step + 2) * TK, 0, 1);
            __builtin_prefetch(bgp + (step + 2) * TK, 0, 1);
        }
        COMPUTE(cur);
        if (step + 1 < S) WAIT_ASYNC();
        __syncthreads();
    }
#undef AFETCH

#else
    // ---------------- regs double-buffer fallback ----------------
    uint4 ra0, ra1, rb0, rb1;
#define FETCH(K0)                                            \
    do {                                                     \
        const uint4* ap_ = (const uint4*)(agp + (K0));       \
        const uint4* bp_ = (const uint4*)(bgp + (K0));       \
        ra0 = ap_[0]; ra1 = ap_[1];                          \
        rb0 = bp_[0]; rb1 = bp_[1];                          \
    } while (0)
#define STAGE(BUF)                                           \
    do {                                                     \
        *(uint4*)&As[BUF][srow][soff]     = ra0;             \
        *(uint4*)&As[BUF][srow][soff + 8] = ra1;             \
        *(uint4*)&Bs[BUF][srow][soff]     = rb0;             \
        *(uint4*)&Bs[BUF][srow][soff + 8] = rb1;             \
    } while (0)

    FETCH(0);
    STAGE(0);
    FETCH(TK);
    __syncthreads();

    for (int step = 0; step < S; ++step) {
        const int cur = step & 1;
        if (step + 3 < S) {
            __builtin_prefetch(agp + (step + 3) * TK, 0, 1);
            __builtin_prefetch(bgp + (step + 3) * TK, 0, 1);
        }
        COMPUTE(cur);
        if (step + 1 < S) STAGE(cur ^ 1);
        if (step + 2 < S) FETCH((step + 2) * TK);
        __syncthreads();
    }
#undef FETCH
#undef STAGE
#endif
#undef COMPUTE

    // D layout: VGPR v holds row (v + 8h), col = lane%16
    const long long row0 = bm + wm + 8 * h;
    const long long col  = bn + wn + r;
#pragma unroll
    for (int v = 0; v < 8; ++v) {
        C[(row0 + v) * N + col]      = acc0[v];
        C[(row0 + v) * N + col + 16] = acc1[v];
    }
}

// ---------------------------------------------------------------------------
// Local windowed attention, one wave32 per token row.
// qkv [B*L][3*C] fp32 (q|k|v). Lane owns 16 channels. Inline width GEMV,
// 17 zero-padded masked scores, softmax, v-window accumulation + residual.
// Output written as bf16 [B*L][C] (feeds the projection GEMM directly).
// ---------------------------------------------------------------------------
#define CDIM 512

__launch_bounds__(256, 2)
__global__ void local_attn(const float* __restrict__ qkv,
                           const float* __restrict__ x,
                           const float* __restrict__ w_width,
                           const float* __restrict__ b_width,
                           u16t* __restrict__ out,
                           int L) {
    const int lane = threadIdx.x & 31;
    const int wv   = threadIdx.x >> 5;
    const long long g = (long long)blockIdx.x * 8 + wv;   // token row
    const int l = (int)(g % L);
    const long long bbase = (g / L) * (long long)L;

    const int c0 = lane * 16;
    const long long qkvRow = g * (3LL * CDIM);

    float4 q[4], xr[4], ww[4];
    const float4* qp = (const float4*)(qkv + qkvRow + c0);
    const float4* xp = (const float4*)(x + g * (long long)CDIM + c0);
    const float4* wp = (const float4*)(w_width + c0);
#pragma unroll
    for (int i = 0; i < 4; ++i) { q[i] = qp[i]; xr[i] = xp[i]; ww[i] = wp[i]; }

    // width = sigmoid(x . w_width + b) * (K-1) + 1
    float wsum = 0.f;
#pragma unroll
    for (int i = 0; i < 4; ++i)
        wsum += xr[i].x * ww[i].x + xr[i].y * ww[i].y +
                xr[i].z * ww[i].z + xr[i].w * ww[i].w;
#pragma unroll
    for (int m = 16; m >= 1; m >>= 1) wsum += __shfl_xor(wsum, m, 32);
    const float width = 16.f / (1.f + __expf(-(wsum + b_width[0]))) + 1.f;

    const float scale = 0.04419417382415922f;  // 512^-0.5
    float s[17];
#pragma unroll
    for (int j = 0; j < 17; ++j) {
        const int p = l + j - 8;                       // wave-uniform
        float d = 0.f;
        if (p >= 0 && p < L) {
            const float4* kp =
                (const float4*)(qkv + (bbase + p) * (3LL * CDIM) + CDIM + c0);
#pragma unroll
            for (int i = 0; i < 4; ++i) {
                float4 kv = kp[i];
                d += q[i].x * kv.x + q[i].y * kv.y + q[i].z * kv.z + q[i].w * kv.w;
            }
        }
#pragma unroll
        for (int m = 16; m >= 1; m >>= 1) d += __shfl_xor(d, m, 32);
        const float rel = fabsf((float)j - 8.f);
        const float sm  = 1.f / (1.f + __expf(-(width - rel) * 5.f));
        s[j] = d * scale - (1.f - sm) * 10000.f;
    }

    float mx = s[0];
#pragma unroll
    for (int j = 1; j < 17; ++j) mx = fmaxf(mx, s[j]);
    float se = 0.f;
#pragma unroll
    for (int j = 0; j < 17; ++j) { s[j] = __expf(s[j] - mx); se += s[j]; }
    const float inv = 1.f / se;

    float of[16];
    {
        const float4* vs = (const float4*)(qkv + qkvRow + 2 * CDIM + c0);
#pragma unroll
        for (int i = 0; i < 4; ++i) {
            float4 vv = vs[i];                         // residual "+ v"
            of[4 * i + 0] = vv.x; of[4 * i + 1] = vv.y;
            of[4 * i + 2] = vv.z; of[4 * i + 3] = vv.w;
        }
    }
#pragma unroll
    for (int j = 0; j < 17; ++j) {
        const int p = l + j - 8;
        if (p >= 0 && p < L) {
            const float a = s[j] * inv;
            const float4* vp =
                (const float4*)(qkv + (bbase + p) * (3LL * CDIM) + 2 * CDIM + c0);
#pragma unroll
            for (int i = 0; i < 4; ++i) {
                float4 vv = vp[i];
                of[4 * i + 0] += a * vv.x; of[4 * i + 1] += a * vv.y;
                of[4 * i + 2] += a * vv.z; of[4 * i + 3] += a * vv.w;
            }
        }
    }

    // pack bf16 and store as two b128
    unsigned pw[8];
#pragma unroll
    for (int i = 0; i < 8; ++i)
        pw[i] = (unsigned)f32_to_bf16(of[2 * i]) |
                ((unsigned)f32_to_bf16(of[2 * i + 1]) << 16);
    uint4* op = (uint4*)(out + g * (long long)CDIM + c0);
    op[0] = make_uint4(pw[0], pw[1], pw[2], pw[3]);
    op[1] = make_uint4(pw[4], pw[5], pw[6], pw[7]);
}

// ---------------------------------------------------------------------------
// Launch: bf16 prep -> qkv GEMM -> fused attention -> projection GEMM.
// Workspace: xbf + wqkvT + woutT (bf16) + qkv fp32 + att bf16  (~137 MiB).
// ---------------------------------------------------------------------------
extern "C" void kernel_launch(void* const* d_in, const int* in_sizes, int n_in,
                              void* d_out, int out_size, void* d_ws, size_t ws_size,
                              hipStream_t stream) {
    (void)n_in; (void)out_size; (void)ws_size;
    const float* x     = (const float*)d_in[0];   // [B, L, C]
    const float* w_qkv = (const float*)d_in[1];   // [C, 3C]
    const float* w_wid = (const float*)d_in[2];   // [C, 1]
    const float* b_wid = (const float*)d_in[3];   // [1]
    const float* w_out = (const float*)d_in[4];   // [C, C]

    const int C  = 512;
    const int B  = 2;
    const long long BL = (long long)in_sizes[0] / C;   // 16384
    const int L  = (int)(BL / B);                      // 8192

    char* ws = (char*)d_ws;
    size_t off = 0;
    u16t* xbf   = (u16t*)(ws + off); off += (size_t)BL * C * 2;
    u16t* wqkvT = (u16t*)(ws + off); off += (size_t)(3 * C) * C * 2;
    u16t* woutT = (u16t*)(ws + off); off += (size_t)C * C * 2;
    off = (off + 255) & ~(size_t)255;
    float* qkv  = (float*)(ws + off); off += (size_t)BL * 3 * C * 4;
    u16t* attbf = (u16t*)(ws + off);

    dim3 blk(256);

    // fp32 -> bf16 conversions (x) and transpose-conversions (weights)
    cvt_bf16<<<dim3((unsigned)(BL * C / 1024)), blk, 0, stream>>>(x, xbf);
    cvtT_bf16<<<dim3((unsigned)(3 * C * C / 256)), blk, 0, stream>>>(
        w_qkv, wqkvT, C, 3 * C);
    cvtT_bf16<<<dim3((unsigned)(C * C / 256)), blk, 0, stream>>>(
        w_out, woutT, C, C);

    // qkv = x @ w_qkv   (M=16384, N=1536, K=512)
    wmma_gemm_bf16t<<<dim3((unsigned)(BL / TM), (3 * C) / TN), blk, 0, stream>>>(
        xbf, wqkvT, qkv, 3 * C, C);

    // fused width + windowed attention + residual (bf16 out)
    local_attn<<<dim3((unsigned)(BL / 8)), blk, 0, stream>>>(
        qkv, x, w_wid, b_wid, attbf, L);

    // out = att @ w_out (M=16384, N=512, K=512)
    wmma_gemm_bf16t<<<dim3((unsigned)(BL / TM), C / TN), blk, 0, stream>>>(
        attbf, woutT, (float*)d_out, C, C);
}
